// TemporalGNN_77292231459487
// MI455X (gfx1250) — matrix operand order repaired and checked by reading, not compiled, not run on hardware
//
#include <hip/hip_runtime.h>
#include <hip/hip_bf16.h>

#define NNODES 50000
#define NEDGES 409600
#define HC 128

typedef __attribute__((ext_vector_type(16))) _Float16 v16h;
typedef __attribute__((ext_vector_type(8)))  float    v8f;

union FragH { v16h v; unsigned int u[8]; uint4 q[2]; };
union PairPack { _Float16 h[2]; unsigned int u; };

__device__ __forceinline__ void atomicMaxF(float* addr, float val) {
    // float max via ordered-int trick (works for mixed signs)
    if (val >= 0.f) atomicMax((int*)addr, __float_as_int(val));
    else            atomicMin((unsigned int*)addr, __float_as_uint(val));
}

__device__ __forceinline__ void atomAddF(float* a, float v) {
    unsafeAtomicAdd(a, v);   // -> global_atomic_add_f32 (no CAS loop)
}

// ---------------------------------------------------------------------------
// u_l = We_enc @ We_l ; c_l = be_enc @ We_l   (rank-1 edge-feature folding)
// ---------------------------------------------------------------------------
__global__ void k_uc(const float* __restrict__ we, const float* __restrict__ be,
                     const float* __restrict__ We1, const float* __restrict__ We2,
                     float* u1, float* c1, float* u2, float* c2) {
    int j = threadIdx.x;  // 0..127
    float a1 = 0.f, b1 = 0.f, a2 = 0.f, b2 = 0.f;
    for (int t = 0; t < 64; ++t) {
        float w1 = We1[t * HC + j], w2 = We2[t * HC + j];
        a1 += we[t] * w1;  b1 += be[t] * w1;
        a2 += we[t] * w2;  b2 += be[t] * w2;
    }
    u1[j] = a1; c1[j] = b1; u2[j] = a2; c2[j] = b2;
}

// ---------------------------------------------------------------------------
// Layer-1 projections (K=6, too skinny for WMMA): q,k,v,skip in one pass
// ---------------------------------------------------------------------------
__global__ void k_proj1(const float* __restrict__ x,
                        const float* __restrict__ Wq, const float* __restrict__ bq,
                        const float* __restrict__ Wk, const float* __restrict__ bk,
                        const float* __restrict__ Wv, const float* __restrict__ bv,
                        const float* __restrict__ Wsk, const float* __restrict__ bs,
                        float* Q, float* K, float* V, float* D) {
    int n = blockIdx.x;
    int j = threadIdx.x;
    float xv[6];
#pragma unroll
    for (int t = 0; t < 6; ++t) xv[t] = x[n * 6 + t];
    float q = bq[j], k = bk[j], v = bv[j], s = bs[j];
#pragma unroll
    for (int t = 0; t < 6; ++t) {
        q = fmaf(xv[t], Wq[t * HC + j], q);
        k = fmaf(xv[t], Wk[t * HC + j], k);
        v = fmaf(xv[t], Wv[t * HC + j], v);
        s = fmaf(xv[t], Wsk[t * HC + j], s);
    }
    size_t o = (size_t)n * HC + j;
    Q[o] = q; K[o] = k; V[o] = v; D[o] = s;
}

// ---------------------------------------------------------------------------
// Pack 4x [128,128] f32 weights into f16 WMMA B-fragments.
// B 32x16 f16 packing used here (k order shared with A): lane L -> col = (L&15); lanes 0-15: vgpr i = K(2i,2i+1),
// lanes 16-31: vgpr i = K(16+2i, 16+2i+1).
// ---------------------------------------------------------------------------
__global__ void k_pack(const float* __restrict__ W0, const float* __restrict__ W1,
                       const float* __restrict__ W2, const float* __restrict__ W3,
                       _Float16* __restrict__ packed) {
    int b = blockIdx.x;          // 128 blocks = 4 weights * 8 ntiles * 4 ksteps
    int lane = threadIdx.x;      // 32
    int w = b >> 5, t = (b >> 2) & 7, ks = b & 3;
    const float* W = (w == 0) ? W0 : (w == 1) ? W1 : (w == 2) ? W2 : W3;
    int col = t * 16 + (lane & 15);
    int khi = (lane >= 16) ? 16 : 0;
    FragH f;
#pragma unroll
    for (int p = 0; p < 8; ++p) {
        int k0 = ks * 32 + khi + 2 * p;
        f.v[2 * p]     = (_Float16)W[k0 * HC + col];
        f.v[2 * p + 1] = (_Float16)W[(k0 + 1) * HC + col];
    }
    uint4* dst = (uint4*)(packed + ((size_t)((w * 8 + t) * 4 + ks) * 32 + lane) * 16);
    dst[0] = f.q[0];
    dst[1] = f.q[1];
}

// ---------------------------------------------------------------------------
// Layer-2 projections: [N,128] @ 4x[128,128] via v_wmma_f32_16x16x32_f16.
// Block = 256 thr (8 waves); block tile = 128 rows; wave tile = 16 rows x 128.
// Activations staged once in LDS as f16 pairs (pitch 66 dwords, no bank dup).
// ---------------------------------------------------------------------------
__global__ void k_gemm4(const float* __restrict__ X, const _Float16* __restrict__ packed,
                        const float* __restrict__ b0, const float* __restrict__ b1,
                        const float* __restrict__ b2, const float* __restrict__ b3,
                        float* o0, float* o1, float* o2, float* o3) {
    __shared__ unsigned int shA[128 * 66];   // 33 KB: 128 rows x 64 f16-pairs (+2 pad)
    int tid = threadIdx.x;
    int rb = blockIdx.x * 128;

    { // stage: 2 threads per row, 64 cols each, packed to f16 pairs
        int r = tid >> 1;
        int hs = tid & 1;
        int grow = rb + r;
        unsigned int* srow = shA + r * 66 + hs * 32;
        if (grow < NNODES) {
            const float* xr = X + (size_t)grow * HC + hs * 64;
#pragma unroll
            for (int p = 0; p < 32; ++p) {
                PairPack pk;
                pk.h[0] = (_Float16)xr[2 * p];
                pk.h[1] = (_Float16)xr[2 * p + 1];
                srow[p] = pk.u;
            }
        } else {
#pragma unroll
            for (int p = 0; p < 32; ++p) srow[p] = 0u;
        }
    }
    __syncthreads();

    int wv = tid >> 5, lane = tid & 31;
    int mlocal = wv * 16 + (lane & 15);
    int kp = (lane >= 16) ? 8 : 0;           // A-frag K pair offset (K units /2)
    const float* biases[4] = {b0, b1, b2, b3};
    float* outs[4] = {o0, o1, o2, o3};

#pragma unroll
    for (int wm = 0; wm < 4; ++wm) {
        v8f acc[8];
        v8f z = {0.f, 0.f, 0.f, 0.f, 0.f, 0.f, 0.f, 0.f};
#pragma unroll
        for (int t = 0; t < 8; ++t) acc[t] = z;
#pragma unroll
        for (int ks = 0; ks < 4; ++ks) {
            FragH a;
#pragma unroll
            for (int p = 0; p < 8; ++p) {
                // A frag k order used here: vgpr p K = koff+2p (koff = 0 or 16), matching B
                int pi = ks * 16 + kp + p;
                a.u[p] = shA[mlocal * 66 + pi];
            }
#pragma unroll
            for (int t = 0; t < 8; ++t) {
                FragH bf;
                const uint4* bp = (const uint4*)(packed +
                    ((size_t)((wm * 8 + t) * 4 + ks) * 32 + lane) * 16);
                bf.q[0] = bp[0];
                bf.q[1] = bp[1];
                acc[t] = __builtin_amdgcn_wmma_f32_16x16x32_f16(
                    false, a.v, false, bf.v, (short)0, acc[t], false, false);
            }
        }
        const float* bias = biases[wm];
        float* O = outs[wm];
#pragma unroll
        for (int t = 0; t < 8; ++t) {
            int col = t * 16 + (lane & 15);
            int rbase = rb + wv * 16 + ((lane >= 16) ? 8 : 0);
            float bc = bias[col];
#pragma unroll
            for (int r = 0; r < 8; ++r) {
                int grow = rbase + r;
                if (grow < NNODES) O[(size_t)grow * HC + col] = acc[t][r] + bc;
            }
        }
    }
}

// ---------------------------------------------------------------------------
// Attention score + segment max. One wave per edge; lane covers 4 channels;
// lanes 0-15 = head 0, 16-31 = head 1.  eh is synthesized: a_i*u + c.
// ---------------------------------------------------------------------------
__global__ void k_alpha(const long long* __restrict__ ei, const float* __restrict__ ea,
                        const float* __restrict__ Q, const float* __restrict__ K,
                        const float* __restrict__ U, const float* __restrict__ Cc,
                        float* alpha, float* m) {
    int i = blockIdx.x * 8 + (threadIdx.x >> 5);
    if (i >= NEDGES) return;
    int lane = threadIdx.x & 31;
    int srcI = (int)ei[i];
    int dstI = (int)ei[NEDGES + i];
    float ai = ea[i];
    int c0 = lane * 4;
    float4 qv = *(const float4*)(Q + (size_t)dstI * HC + c0);
    float4 kv = *(const float4*)(K + (size_t)srcI * HC + c0);
    float4 uv = *(const float4*)(U + c0);
    float4 cv = *(const float4*)(Cc + c0);
    float p = qv.x * (kv.x + ai * uv.x + cv.x)
            + qv.y * (kv.y + ai * uv.y + cv.y)
            + qv.z * (kv.z + ai * uv.z + cv.z)
            + qv.w * (kv.w + ai * uv.w + cv.w);
    p += __shfl_xor(p, 1);
    p += __shfl_xor(p, 2);
    p += __shfl_xor(p, 4);
    p += __shfl_xor(p, 8);
    if ((lane & 15) == 0) {
        int h = lane >> 4;
        float al = p * 0.125f;   // 1/sqrt(64)
        alpha[(size_t)i * 2 + h] = al;
        atomicMaxF(&m[(size_t)dstI * 2 + h], al);
    }
}

// exp(alpha - m[dst]) and segment sum
__global__ void k_expsum(const long long* __restrict__ ei, float* alpha,
                         const float* __restrict__ m, float* s) {
    int idx = blockIdx.x * 256 + threadIdx.x;
    if (idx >= NEDGES * 2) return;
    int i = idx >> 1, h = idx & 1;
    int dstI = (int)ei[NEDGES + i];
    float ex = __expf(alpha[idx] - m[(size_t)dstI * 2 + h]);
    alpha[idx] = ex;
    atomAddF(&s[(size_t)dstI * 2 + h], ex);
}

// weighted message scatter: D[dst] += (v[src] + eh) * softmax
__global__ void k_message(const long long* __restrict__ ei, const float* __restrict__ ea,
                          const float* __restrict__ V, const float* __restrict__ U,
                          const float* __restrict__ Cc, const float* __restrict__ alpha,
                          const float* __restrict__ s, float* D) {
    int i = blockIdx.x * 8 + (threadIdx.x >> 5);
    if (i >= NEDGES) return;
    int lane = threadIdx.x & 31;
    int srcI = (int)ei[i];
    int dstI = (int)ei[NEDGES + i];
    float ai = ea[i];
    int c0 = lane * 4;
    int h = lane >> 4;
    float coef = alpha[(size_t)i * 2 + h] / (s[(size_t)dstI * 2 + h] + 1e-16f);
    float4 vv = *(const float4*)(V + (size_t)srcI * HC + c0);
    float4 uv = *(const float4*)(U + c0);
    float4 cv = *(const float4*)(Cc + c0);
    float* dp = D + (size_t)dstI * HC + c0;
    atomAddF(dp + 0, (vv.x + ai * uv.x + cv.x) * coef);
    atomAddF(dp + 1, (vv.y + ai * uv.y + cv.y) * coef);
    atomAddF(dp + 2, (vv.z + ai * uv.z + cv.z) * coef);
    atomAddF(dp + 3, (vv.w + ai * uv.w + cv.w) * coef);
}

__global__ void k_initms(float* m, float* s) {
    int idx = blockIdx.x * 256 + threadIdx.x;
    if (idx < NNODES * 2) { m[idx] = -3.402823466e38f; s[idx] = 0.f; }
}

__global__ void k_relu(const float* __restrict__ in, float* out, int n) {
    int idx = blockIdx.x * 256 + threadIdx.x;
    if (idx < n) out[idx] = fmaxf(in[idx], 0.f);
}

// final fc: [N,128] @ [128,1] + b, one wave per node
__global__ void k_fc(const float* __restrict__ Hf, const float* __restrict__ Wfc,
                     const float* __restrict__ bfc, float* out) {
    int n = blockIdx.x * 8 + (threadIdx.x >> 5);
    if (n >= NNODES) return;
    int lane = threadIdx.x & 31;
    float4 hv = *(const float4*)(Hf + (size_t)n * HC + lane * 4);
    float4 wv = *(const float4*)(Wfc + lane * 4);
    float p = hv.x * wv.x + hv.y * wv.y + hv.z * wv.z + hv.w * wv.w;
    p += __shfl_xor(p, 1);
    p += __shfl_xor(p, 2);
    p += __shfl_xor(p, 4);
    p += __shfl_xor(p, 8);
    p += __shfl_xor(p, 16);
    if (lane == 0) out[n] = p + bfc[0];
}

extern "C" void kernel_launch(void* const* d_in, const int* in_sizes, int n_in,
                              void* d_out, int out_size, void* d_ws, size_t ws_size,
                              hipStream_t stream) {
    const float*     x      = (const float*)d_in[0];
    const long long* ei     = (const long long*)d_in[1];
    const float*     ea     = (const float*)d_in[2];
    const float*     We_enc = (const float*)d_in[3];
    const float*     be_enc = (const float*)d_in[4];
    const float* Wq1 = (const float*)d_in[5];  const float* bq1 = (const float*)d_in[6];
    const float* Wk1 = (const float*)d_in[7];  const float* bk1 = (const float*)d_in[8];
    const float* Wv1 = (const float*)d_in[9];  const float* bv1 = (const float*)d_in[10];
    const float* We1 = (const float*)d_in[11];
    const float* Ws1 = (const float*)d_in[12]; const float* bs1 = (const float*)d_in[13];
    const float* Wq2 = (const float*)d_in[14]; const float* bq2 = (const float*)d_in[15];
    const float* Wk2 = (const float*)d_in[16]; const float* bk2 = (const float*)d_in[17];
    const float* Wv2 = (const float*)d_in[18]; const float* bv2 = (const float*)d_in[19];
    const float* We2 = (const float*)d_in[20];
    const float* Ws2 = (const float*)d_in[21]; const float* bs2 = (const float*)d_in[22];
    const float* Wfc = (const float*)d_in[23]; const float* bfc = (const float*)d_in[24];

    float* ws = (float*)d_ws;
    size_t off = 0;
    const size_t NF = (size_t)NNODES * HC;
    float* Qb  = ws + off; off += NF;
    float* Kb  = ws + off; off += NF;
    float* Vb  = ws + off; off += NF;
    float* Db  = ws + off; off += NF;
    float* Hb  = ws + off; off += NF;
    float* mB  = ws + off; off += (size_t)NNODES * 2;
    float* sB  = ws + off; off += (size_t)NNODES * 2;
    float* alB = ws + off; off += (size_t)NEDGES * 2;
    float* u1  = ws + off; off += 128;
    float* c1  = ws + off; off += 128;
    float* u2  = ws + off; off += 128;
    float* c2  = ws + off; off += 128;
    _Float16* packed = (_Float16*)(ws + off); off += 32768;  // 64K halves

    const int EB  = NEDGES / 8;                 // wave-per-edge blocks (51200)
    const int NB  = (NNODES + 127) / 128;       // gemm blocks (391)
    const int MSB = (NNODES * 2 + 255) / 256;   // init m/s
    const int RB  = ((int)NF + 255) / 256;      // relu
    const int E2B = (NEDGES * 2 + 255) / 256;   // expsum
    const int FCB = (NNODES + 7) / 8;           // fc

    // fold edge-feature path into rank-1 vectors; pack layer-2 weights to f16 frags
    k_uc<<<1, 128, 0, stream>>>(We_enc, be_enc, We1, We2, u1, c1, u2, c2);
    k_pack<<<128, 32, 0, stream>>>(Wq2, Wk2, Wv2, Ws2, packed);

    // ----- layer 1 -----
    k_proj1<<<NNODES, 128, 0, stream>>>(x, Wq1, bq1, Wk1, bk1, Wv1, bv1, Ws1, bs1,
                                        Qb, Kb, Vb, Db);
    k_initms<<<MSB, 256, 0, stream>>>(mB, sB);
    k_alpha<<<EB, 256, 0, stream>>>(ei, ea, Qb, Kb, u1, c1, alB, mB);
    k_expsum<<<E2B, 256, 0, stream>>>(ei, alB, mB, sB);
    k_message<<<EB, 256, 0, stream>>>(ei, ea, Vb, u1, c1, alB, sB, Db);
    k_relu<<<RB, 256, 0, stream>>>(Db, Hb, (int)NF);

    // ----- layer 2 -----
    k_gemm4<<<NB, 256, 0, stream>>>(Hb, packed, bq2, bk2, bv2, bs2, Qb, Kb, Vb, Db);
    k_initms<<<MSB, 256, 0, stream>>>(mB, sB);
    k_alpha<<<EB, 256, 0, stream>>>(ei, ea, Qb, Kb, u2, c2, alB, mB);
    k_expsum<<<E2B, 256, 0, stream>>>(ei, alB, mB, sB);
    k_message<<<EB, 256, 0, stream>>>(ei, ea, Vb, u2, c2, alB, sB, Db);
    k_relu<<<RB, 256, 0, stream>>>(Db, Db, (int)NF);

    // ----- fc head -----
    k_fc<<<FCB, 256, 0, stream>>>(Db, Wfc, bfc, (float*)d_out);
}
